// SMDScaleAttention_29308856827968
// MI455X (gfx1250) — matrix operand order repaired and checked by reading, not compiled, not run on hardware
//
#include <hip/hip_runtime.h>
#include <hip/hip_bf16.h>
#include <math.h>

// ---------------------------------------------------------------------------
// SMD-scale attention for MI455X (gfx1250, wave32, WMMA + TDM).
// GEMMs: fp16 operands, fp32 accumulate via v_wmma_f32_16x16x32_f16, with
// operand tiles staged LDS-side by the Tensor Data Mover (tensor_load_to_lds,
// TENSORcnt-synchronized). Iterative SMD loop: fp32 VALU elementwise kernels.
// ---------------------------------------------------------------------------

typedef _Float16 h16;
typedef __attribute__((ext_vector_type(8)))  _Float16 h16x8;
typedef __attribute__((ext_vector_type(16))) _Float16 h16x16;
typedef __attribute__((ext_vector_type(8)))  float    f32x8;
typedef __attribute__((ext_vector_type(4)))  unsigned u32x4;
typedef __attribute__((ext_vector_type(8)))  int      i32x8;
typedef __attribute__((ext_vector_type(4)))  int      i32x4;

#define WG 256
#define TDM_PITCH 40   // h16 per LDS tile row: 32 data + 4-dword TDM pad (80 B)

// ---------------- conversion / repack kernels ----------------

__global__ void cvt_f32_to_h16(const float* __restrict__ in, h16* __restrict__ out, size_t n) {
  for (size_t i = (size_t)blockIdx.x * blockDim.x + threadIdx.x; i < n;
       i += (size_t)gridDim.x * blockDim.x)
    out[i] = (h16)in[i];
}

// in: R x C fp32 row-major  ->  out: C x R fp16 row-major (transpose)
__global__ void cvt_t_f32_to_h16(const float* __restrict__ in, h16* __restrict__ out,
                                 int R, int C) {
  size_t total = (size_t)R * C;
  for (size_t i = (size_t)blockIdx.x * blockDim.x + threadIdx.x; i < total;
       i += (size_t)gridDim.x * blockDim.x) {
    int r = (int)(i / C);
    int c = (int)(i % C);
    out[(size_t)c * R + r] = (h16)in[i];
  }
}

// in: (Hh, Fd, dk) fp32  ->  out: (Hh*dk, Fd) fp16  (B-transposed for NT gemm)
__global__ void repack_smd_w(const float* __restrict__ in, h16* __restrict__ out,
                             int Hh, int Fd, int dk) {
  size_t total = (size_t)Hh * Fd * dk;
  for (size_t i = (size_t)blockIdx.x * blockDim.x + threadIdx.x; i < total;
       i += (size_t)gridDim.x * blockDim.x) {
    int d = (int)(i % dk);
    size_t t = i / dk;
    int f = (int)(t % Fd);
    int h = (int)(t / Fd);
    out[(size_t)(h * dk + d) * Fd + f] = (h16)in[i];
  }
}

// ---------------- TDM: DMA a 128-row x 32-col h16 tile into LDS ----------------
// D# packing per CDNA5 ISA. Rows are padded by the
// TDM (pad after every 16 dwords, +4 dwords) -> 80B LDS pitch, bank-friendly.
// Rows beyond rows_valid are zero-filled by the TDM's OOB handling.

__device__ __forceinline__ void tdm_load_tile_h16(unsigned lds_off, const h16* gptr,
                                                  int rows_valid, int stride_elems) {
  unsigned long long ga = (unsigned long long)(uintptr_t)gptr;
  u32x4 g0;
  g0.x = 1u;                                   // count=1, is_restore=0
  g0.y = lds_off;                              // lds_addr (bytes)
  g0.z = (unsigned)ga;                         // global_addr[31:0]
  g0.w = (unsigned)(ga >> 32) | (2u << 30);    // global_addr[56:32] | type=2
  i32x8 g1;
  unsigned w0 = (1u << 16)                     // data_size = 2 bytes
              | (1u << 20)                     // pad_enable
              | (3u << 22)                     // pad_interval: every 16 dwords
              | (3u << 25);                    // pad_amount: 4 dwords (16 B)
  unsigned td0 = 32u;                          // tensor_dim0 (cols from tile start)
  unsigned td1 = (unsigned)rows_valid;         // tensor_dim1 (valid rows)
  g1[0] = (int)w0;
  g1[1] = (int)((td0 & 0xFFFFu) << 16);                    // tensor_dim0[15:0]
  g1[2] = (int)((td0 >> 16) | ((td1 & 0xFFFFu) << 16));    // td0[31:16] | td1[15:0]
  g1[3] = (int)((td1 >> 16) | (32u << 16));                // td1[31:16] | tile_dim0=32
  g1[4] = (int)128u;                                       // tile_dim1=128, tile_dim2=0
  g1[5] = (int)(unsigned)stride_elems;                     // tensor_dim0_stride[31:0]
  g1[6] = 0;                                               // stride msbs / dim1_stride
  g1[7] = 0;
  i32x4 z4 = {0, 0, 0, 0};                     // groups 2/3: dims zero = unused
  i32x8 z8 = {0, 0, 0, 0, 0, 0, 0, 0};
  __builtin_amdgcn_tensor_load_to_lds(g0, g1, z4, z4, z8, 0);
}

// ---------------- WMMA GEMM: C(f32, MxN) = scale * A(h16, MxK) * Bt(h16, NxK)^T ----
// Block: 256 thr = 8 waves; block tile 128(M) x 128(N); wave tile 64x32
// (4x2 WMMA 16x16 tiles). A/B tiles staged in LDS by the TDM each K-step.

__global__ __launch_bounds__(256)
void gemm_nt_wmma(const h16* __restrict__ A, const h16* __restrict__ Bt,
                  float* __restrict__ C, int M, int Nn, int K,
                  int lda, int ldb, int ldc, float scale) {
  __shared__ __align__(16) h16 As[128 * TDM_PITCH];
  __shared__ __align__(16) h16 Bs[128 * TDM_PITCH];

  const int lane = threadIdx.x & 31;
  const int wave = threadIdx.x >> 5;
  const int l16  = lane & 15;
  const int hi   = lane >> 4;       // 0 or 1 (half-wave)
  const int wm   = wave & 1;        // 2 waves along M (64 rows each)
  const int wn   = wave >> 1;       // 4 waves along N (32 cols each)
  const int m0   = wm * 64;
  const int n0   = wn * 32;
  const int bm   = blockIdx.x * 128;
  const int bn   = blockIdx.y * 128;
  const int rowsB = (Nn - bn) < 128 ? (Nn - bn) : 128;

  f32x8 acc[4][2] = {};

  for (int k0 = 0; k0 < K; k0 += 32) {
    __syncthreads();                 // previous iteration done reading LDS
    if (threadIdx.x < 32) {          // wave 0 drives the TDM
      tdm_load_tile_h16((unsigned)(uintptr_t)As, A + (size_t)bm * lda + k0, 128, lda);
      tdm_load_tile_h16((unsigned)(uintptr_t)Bs, Bt + (size_t)bn * ldb + k0, rowsB, ldb);
      __builtin_amdgcn_s_wait_tensorcnt(0);
    }
    __syncthreads();                 // tiles visible to all waves

    h16x16 b[2];
#pragma unroll
    for (int j = 0; j < 2; ++j) {
      // B 32x16 fragment: lanes 0-15 col N=l16 K=0..15, lanes 16-31 K=16..31
      const h16* bp = Bs + (n0 + j * 16 + l16) * TDM_PITCH + hi * 16;
      h16x8 blo = *(const h16x8*)bp;
      h16x8 bhi = *(const h16x8*)(bp + 8);
      b[j] = __builtin_shufflevector(blo, bhi, 0, 1, 2, 3, 4, 5, 6, 7,
                                     8, 9, 10, 11, 12, 13, 14, 15);
    }
#pragma unroll
    for (int i = 0; i < 4; ++i) {
      // A 16x32 fragment: lanes 0-15 K=0..15, lanes 16-31 K=16..31
      const h16* ap = As + (m0 + i * 16 + l16) * TDM_PITCH + hi * 16;
      h16x8 alo = *(const h16x8*)ap;
      h16x8 ahi = *(const h16x8*)(ap + 8);
      h16x16 a = __builtin_shufflevector(alo, ahi, 0, 1, 2, 3, 4, 5, 6, 7,
                                         8, 9, 10, 11, 12, 13, 14, 15);
#pragma unroll
      for (int j = 0; j < 2; ++j)
        acc[i][j] = __builtin_amdgcn_wmma_f32_16x16x32_f16(
            false, a, false, b[j], (short)0, acc[i][j], false, false);
    }
  }

#pragma unroll
  for (int i = 0; i < 4; ++i)
#pragma unroll
    for (int j = 0; j < 2; ++j)
#pragma unroll
      for (int v = 0; v < 8; ++v) {
        int row = bm + m0 + i * 16 + hi * 8 + v;   // C layout: VGPR v, half-wave hi
        int col = bn + n0 + j * 16 + l16;
        if (col < Nn)
          C[(size_t)row * ldc + col] = acc[i][j][v] * scale;
      }
}

// ---------------- RoPE (fp32 in, fp16 out), layout N x (Hh*64) ----------------

__global__ void rope_to_h16(const float* __restrict__ in, h16* __restrict__ out,
                            int N, int Hh) {
  const int dk = 64, half = 32;
  int total = N * Hh * half;
  int i = blockIdx.x * blockDim.x + threadIdx.x;
  if (i >= total) return;
  int p = i % half;
  int t = i / half;
  int h = t % Hh;
  int n = t / Hh;
  float inv_freq = __powf(10000.f, -(float)(2 * p) / (float)dk);
  float ang = (float)n * inv_freq;
  float s, c;
  __sincosf(ang, &s, &c);
  size_t base = (size_t)n * Hh * dk + (size_t)h * dk;
  float x1 = in[base + p];
  float x2 = in[base + p + half];
  out[base + p]        = (h16)(x1 * c - x2 * s);
  out[base + p + half] = (h16)(x2 * c + x1 * s);
}

// ---------------- charge init: sigmoid(features . charge_w[h] + b[h]) -----------

__global__ void charge_init(const float* __restrict__ feat, const float* __restrict__ cw,
                            const float* __restrict__ cb, float* __restrict__ charge,
                            int N, int Fd, int Hh) {
  int warp = (blockIdx.x * blockDim.x + threadIdx.x) >> 5;
  int lane = threadIdx.x & 31;
  if (warp >= Hh * N) return;
  int h = warp / N, n = warp % N;
  const float* f = feat + (size_t)n * Fd;
  const float* w = cw + (size_t)h * Fd;
  float acc = 0.f;
  for (int i = lane; i < Fd; i += 32) acc += f[i] * w[i];
  for (int off = 16; off > 0; off >>= 1) acc += __shfl_down(acc, off, 32);
  if (lane == 0) charge[(size_t)h * N + n] = 1.f / (1.f + __expf(-(acc + cb[h])));
}

// ---------------- row softmax (fp32 out / fp16 out) ----------------

__global__ void softmax_rows_f32(const float* __restrict__ in, float* __restrict__ out,
                                 int ncols) {
  __shared__ float red[WG];
  size_t row = blockIdx.x;
  const float* r = in + row * (size_t)ncols;
  float* o = out + row * (size_t)ncols;
  int tid = threadIdx.x;
  float m = -3.0e38f;
  for (int c = tid; c < ncols; c += WG) m = fmaxf(m, r[c]);
  red[tid] = m; __syncthreads();
  for (int s = WG / 2; s > 0; s >>= 1) {
    if (tid < s) red[tid] = fmaxf(red[tid], red[tid + s]);
    __syncthreads();
  }
  m = red[0]; __syncthreads();
  float sum = 0.f;
  for (int c = tid; c < ncols; c += WG) { float e = __expf(r[c] - m); o[c] = e; sum += e; }
  red[tid] = sum; __syncthreads();
  for (int s = WG / 2; s > 0; s >>= 1) {
    if (tid < s) red[tid] += red[tid + s];
    __syncthreads();
  }
  float inv = 1.f / red[0];
  for (int c = tid; c < ncols; c += WG) o[c] *= inv;
}

__global__ void softmax_rows_h16(const float* __restrict__ in, h16* __restrict__ out,
                                 int ncols) {
  __shared__ float red[WG];
  size_t row = blockIdx.x;
  const float* r = in + row * (size_t)ncols;
  h16* o = out + row * (size_t)ncols;
  int tid = threadIdx.x;
  float m = -3.0e38f;
  for (int c = tid; c < ncols; c += WG) m = fmaxf(m, r[c]);
  red[tid] = m; __syncthreads();
  for (int s = WG / 2; s > 0; s >>= 1) {
    if (tid < s) red[tid] = fmaxf(red[tid], red[tid + s]);
    __syncthreads();
  }
  m = red[0]; __syncthreads();
  float sum = 0.f;
  for (int c = tid; c < ncols; c += WG) { float e = __expf(r[c] - m); o[c] = (h16)e; sum += e; }
  red[tid] = sum; __syncthreads();
  for (int s = WG / 2; s > 0; s >>= 1) {
    if (tid < s) red[tid] += red[tid + s];
    __syncthreads();
  }
  float inv = 1.f / red[0];
  for (int c = tid; c < ncols; c += WG) o[c] = (h16)((float)o[c] * inv);
}

// ---------------- column sums: recv[h][m] = sum_n attn[h][n][m] ----------------

__global__ void colsum(const float* __restrict__ tmp, float* __restrict__ recv,
                       int N, int Hh) {
  int i = blockIdx.x * blockDim.x + threadIdx.x;
  if (i >= Hh * N) return;
  int h = i / N, m = i % N;
  const float* base = tmp + (size_t)h * N * N + m;
  float s = 0.f;
  for (int n = 0; n < N; ++n) s += base[(size_t)n * N];
  recv[i] = s;
}

__global__ void charge_update(float* __restrict__ charge, const float* __restrict__ recv,
                              const float* __restrict__ cd, int N, int Hh) {
  int i = blockIdx.x * blockDim.x + threadIdx.x;
  if (i >= Hh * N) return;
  int h = i / N;
  float sg = 1.f / (1.f + __expf(-(recv[i] - 1.f)));
  charge[i] *= (1.f - cd[h] * sg);
}

// logits += ss[h] * compat * charge[h][n] * charge[h][m]
__global__ void logits_update(float* __restrict__ logits, const float* __restrict__ compat,
                              const float* __restrict__ charge, const float* __restrict__ ss,
                              int N, int Hh) {
  size_t total = (size_t)Hh * N * N;
  for (size_t i = (size_t)blockIdx.x * blockDim.x + threadIdx.x; i < total;
       i += (size_t)gridDim.x * blockDim.x) {
    int m = (int)(i % N);
    size_t t = i / N;
    int n = (int)(t % N);
    int h = (int)(t / N);
    logits[i] += ss[h] * compat[i] * charge[(size_t)h * N + n] * charge[(size_t)h * N + m];
  }
}

// ---------------------------------------------------------------------------

extern "C" void kernel_launch(void* const* d_in, const int* in_sizes, int n_in,
                              void* d_out, int out_size, void* d_ws, size_t ws_size,
                              hipStream_t stream) {
  const int N = 2048, D = 1024, Fd = 1024, dk = 64;
  const int NSMD = 2, NSTD = 6, H = 8;
  const int SMD_E = NSMD * dk;   // 128
  const int STD_E = NSTD * dk;   // 384
  const int HD    = H * dk;      // 512
  const int STEPS = 4;           // setup_inputs n_steps (host-side constant)
  const float SCALE = 0.125f;    // 1/sqrt(64)

  const float* x        = (const float*)d_in[0];
  const float* features = (const float*)d_in[1];
  const float* Wq_smd   = (const float*)d_in[2];
  const float* Wk_smd   = (const float*)d_in[3];
  const float* charge_w = (const float*)d_in[4];
  const float* charge_b = (const float*)d_in[5];
  const float* step_sz  = (const float*)d_in[6];
  const float* chg_dec  = (const float*)d_in[7];
  const float* Wq_std   = (const float*)d_in[8];
  const float* Wk_std   = (const float*)d_in[9];
  const float* Wv       = (const float*)d_in[10];
  const float* Wo       = (const float*)d_in[11];
  float* out = (float*)d_out;

  // ---- workspace carve-out ----
  char* ws = (char*)d_ws;
  size_t off = 0;
  auto alloc = [&](size_t bytes) -> void* {
    off = (off + 255) & ~(size_t)255;
    void* p = ws + off;
    off += bytes;
    return p;
  };

  h16* fb      = (h16*)alloc((size_t)N * Fd * 2);
  h16* xb      = (h16*)alloc((size_t)N * D * 2);
  h16* WqS_t   = (h16*)alloc((size_t)SMD_E * Fd * 2);
  h16* WkS_t   = (h16*)alloc((size_t)SMD_E * Fd * 2);
  h16* Wqs_t   = (h16*)alloc((size_t)STD_E * D * 2);
  h16* Wks_t   = (h16*)alloc((size_t)STD_E * D * 2);
  h16* Wv_t    = (h16*)alloc((size_t)HD * D * 2);
  h16* Wo_t    = (h16*)alloc((size_t)D * HD * 2);
  float* Qsmd_f = (float*)alloc((size_t)N * SMD_E * 4);
  float* Ksmd_f = (float*)alloc((size_t)N * SMD_E * 4);
  float* Qs_f   = (float*)alloc((size_t)N * STD_E * 4);
  float* Ks_f   = (float*)alloc((size_t)N * STD_E * 4);
  float* V_f    = (float*)alloc((size_t)N * HD * 4);
  h16* Qsmd_h  = (h16*)alloc((size_t)N * SMD_E * 2);
  h16* Ksmd_h  = (h16*)alloc((size_t)N * SMD_E * 2);
  h16* Qs_h    = (h16*)alloc((size_t)N * STD_E * 2);
  h16* Ks_h    = (h16*)alloc((size_t)N * STD_E * 2);
  h16* Vt      = (h16*)alloc((size_t)HD * N * 2);
  float* compat = (float*)alloc((size_t)NSMD * N * N * 4);
  float* logits = (float*)alloc((size_t)NSMD * N * N * 4);
  float* tmp    = (float*)alloc((size_t)NSMD * N * N * 4);
  h16* attn_h  = (h16*)alloc((size_t)N * N * 2);
  float* charge   = (float*)alloc((size_t)NSMD * N * 4);
  float* received = (float*)alloc((size_t)NSMD * N * 4);
  float* outheads   = (float*)alloc((size_t)N * HD * 4);
  h16* outheads_h = (h16*)alloc((size_t)N * HD * 2);
  (void)ws_size; (void)in_sizes; (void)n_in; (void)out_size;

  auto gemm = [&](const h16* A, const h16* Bt, float* C, int M, int Nn, int K,
                  int lda, int ldb, int ldc, float sc) {
    dim3 g((M + 127) / 128, (Nn + 127) / 128);
    gemm_nt_wmma<<<g, dim3(256), 0, stream>>>(A, Bt, C, M, Nn, K, lda, ldb, ldc, sc);
  };

  // ---- precision conversions / weight transposes ----
  cvt_f32_to_h16<<<1024, WG, 0, stream>>>(features, fb, (size_t)N * Fd);
  cvt_f32_to_h16<<<1024, WG, 0, stream>>>(x, xb, (size_t)N * D);
  repack_smd_w<<<512, WG, 0, stream>>>(Wq_smd, WqS_t, NSMD, Fd, dk);
  repack_smd_w<<<512, WG, 0, stream>>>(Wk_smd, WkS_t, NSMD, Fd, dk);
  cvt_t_f32_to_h16<<<1024, WG, 0, stream>>>(Wq_std, Wqs_t, D, STD_E);
  cvt_t_f32_to_h16<<<1024, WG, 0, stream>>>(Wk_std, Wks_t, D, STD_E);
  cvt_t_f32_to_h16<<<1024, WG, 0, stream>>>(Wv, Wv_t, D, HD);
  cvt_t_f32_to_h16<<<1024, WG, 0, stream>>>(Wo, Wo_t, HD, D);

  // ---- projections (WMMA) ----
  gemm(fb, WqS_t, Qsmd_f, N, SMD_E, Fd, Fd, Fd, SMD_E, 1.f);
  gemm(fb, WkS_t, Ksmd_f, N, SMD_E, Fd, Fd, Fd, SMD_E, 1.f);
  gemm(xb, Wqs_t, Qs_f, N, STD_E, D, D, D, STD_E, 1.f);
  gemm(xb, Wks_t, Ks_f, N, STD_E, D, D, D, STD_E, 1.f);
  gemm(xb, Wv_t, V_f, N, HD, D, D, D, HD, 1.f);

  // ---- RoPE + fp16 ----
  int rsm = N * NSMD * 32, rst = N * NSTD * 32;
  rope_to_h16<<<(rsm + WG - 1) / WG, WG, 0, stream>>>(Qsmd_f, Qsmd_h, N, NSMD);
  rope_to_h16<<<(rsm + WG - 1) / WG, WG, 0, stream>>>(Ksmd_f, Ksmd_h, N, NSMD);
  rope_to_h16<<<(rst + WG - 1) / WG, WG, 0, stream>>>(Qs_f, Qs_h, N, NSTD);
  rope_to_h16<<<(rst + WG - 1) / WG, WG, 0, stream>>>(Ks_f, Ks_h, N, NSTD);
  cvt_t_f32_to_h16<<<2048, WG, 0, stream>>>(V_f, Vt, N, HD);   // V^T: HD x N

  // ---- charge ----
  charge_init<<<(NSMD * N * 32 + WG - 1) / WG, WG, 0, stream>>>(
      features, charge_w, charge_b, charge, N, Fd, NSMD);

  // ---- SMD compat = Q K^T * scale ----
  for (int h = 0; h < NSMD; ++h)
    gemm(Qsmd_h + h * dk, Ksmd_h + h * dk, compat + (size_t)h * N * N,
         N, N, dk, SMD_E, SMD_E, N, SCALE);

  (void)hipMemcpyAsync(logits, compat, (size_t)NSMD * N * N * 4,
                       hipMemcpyDeviceToDevice, stream);

  // ---- iterative charge-softmax loop ----
  for (int s = 0; s < STEPS; ++s) {
    softmax_rows_f32<<<NSMD * N, WG, 0, stream>>>(logits, tmp, N);
    colsum<<<(NSMD * N + WG - 1) / WG, WG, 0, stream>>>(tmp, received, N, NSMD);
    charge_update<<<(NSMD * N + WG - 1) / WG, WG, 0, stream>>>(charge, received,
                                                               chg_dec, N, NSMD);
    logits_update<<<8192, WG, 0, stream>>>(logits, compat, charge, step_sz, N, NSMD);
  }

  // ---- per-head: softmax -> fp16 probs -> attn @ V (WMMA) ----
  for (int h = 0; h < H; ++h) {
    const float* src;
    if (h < NSMD) {
      src = logits + (size_t)h * N * N;
    } else {
      int hs = h - NSMD;
      gemm(Qs_h + hs * dk, Ks_h + hs * dk, tmp, N, N, dk, STD_E, STD_E, N, SCALE);
      src = tmp;
    }
    softmax_rows_h16<<<N, WG, 0, stream>>>(src, attn_h, N);
    gemm(attn_h, Vt + (size_t)h * dk * N, outheads + h * dk,
         N, dk, N, N, N, HD, 1.f);
  }

  // ---- output projection ----
  cvt_f32_to_h16<<<1024, WG, 0, stream>>>(outheads, outheads_h, (size_t)N * HD);
  gemm(outheads_h, Wo_t, out, N, D, HD, HD, HD, D, 1.f);
}
